// LongConv_19842748908256
// MI455X (gfx1250) — matrix operand order repaired and checked by reading, not compiled, not run on hardware
//
#include <hip/hip_runtime.h>

// ---------------------------------------------------------------------------
// LongConv on MI455X (gfx1250, wave32, WMMA):
//   conv as Toeplitz bf16 WMMA GEMM (M=16 batches) with exact-sparsity skip,
//   4 l-tiles (4 accumulators) per wave sharing one A operand build,
//   branch-free inner loop via per-channel zero guard tiles,
//   fused skip+GELU, then WMMA channel-mix GEMM with fused bias+SiLU.
// ---------------------------------------------------------------------------

typedef __attribute__((ext_vector_type(16))) __bf16 v16bf;
typedef __attribute__((ext_vector_type(8)))  float  v8f;

#define B_    16
#define H_    512
#define L_    4096
#define LAM   0.001f
#define NT16  (L_ / 16)        // 256 distinct Toeplitz shifts d/16
#define NTG   (NT16 + 2)       // + 2 zero guard tiles (d16 = -2, -1) per channel

// workspace layout (bytes)
#define OFF_KSQ   ((size_t)0)
#define SZ_KSQ    ((size_t)H_ * L_ * 2)                 //   4 MB  squashed k, bf16
#define OFF_JMAX  (OFF_KSQ + SZ_KSQ)
#define SZ_JMAX   ((size_t)4096)                        //   jmax[h]
#define OFF_BTOEP (OFF_JMAX + SZ_JMAX)
#define SZ_BTOEP  ((size_t)H_ * NTG * 512 * 2)          // 129 MB  Toeplitz operand images
#define OFF_WBF   (OFF_BTOEP + SZ_BTOEP)
#define SZ_WBF    ((size_t)H_ * H_ * 2)                 // 0.5 MB  Wout bf16
#define OFF_YACT  (OFF_WBF + SZ_WBF)                    //  64 MB  gelu activations bf16

// ---------------------------------------------------------------------------
// K1: squash kernel -> bf16, and per-channel max nonzero tap index jmax[h].
// ---------------------------------------------------------------------------
__global__ __launch_bounds__(256)
void prep_kernel(const float* __restrict__ kern, __bf16* __restrict__ ksq,
                 int* __restrict__ jmax)
{
    __shared__ int red;
    const int h = blockIdx.x;
    if (threadIdx.x == 0) red = -1;
    __syncthreads();
    int mymax = -1;
    for (int j = threadIdx.x; j < L_; j += 256) {
        const float kv = kern[(size_t)h * L_ + j];
        const float m  = fabsf(kv) - LAM;
        float s = 0.0f;
        if (m > 0.0f) s = (kv > 0.0f) ? m : -m;     // relu(|k|-lam)*sign(k)
        ksq[(size_t)h * L_ + j] = (__bf16)s;
        if (s != 0.0f) mymax = j;
    }
    atomicMax(&red, mymax);
    __syncthreads();
    if (threadIdx.x == 0) jmax[h] = red;
}

// ---------------------------------------------------------------------------
// K2: build Toeplitz B-operand images in exact WMMA lane order.
// B(32x16): lane l -> column n=l%16, element e -> K = 8*(l/16)+e (e<8), 16+8*(l/16)+(e-8) (e>=8).
// B[K][n] = ksq[h][d + n - K] (0 outside [0,L)), d = d16*16, d16 in [-2, 256).
// d16 = -2, -1 are all-zero guard tiles so the conv loop needs no edge branch.
// Stored as btoep[(h*NTG + 2 + d16)*32 + lane][16]: one contiguous 32B chunk
// per lane per tile; the 4 tiles a wave needs per chunk are 4KB contiguous.
// ---------------------------------------------------------------------------
__global__ __launch_bounds__(256)
void toep_kernel(const __bf16* __restrict__ ksq, __bf16* __restrict__ btoep)
{
    const int bx = blockIdx.x;            // h*NTG + (2 + d16)
    const int h  = bx / NTG;
    const int d  = (bx - h * NTG - 2) << 4;   // -32, -16, 0, ..., 4080
    const __bf16* kr = ksq + (size_t)h * L_;
    for (int i = threadIdx.x; i < 512; i += 256) {
        const int lane = i >> 4, e = i & 15;
        const int n = lane & 15, half = lane >> 4;
        const int K = (e < 8) ? ((half << 3) + e) : (16 + (half << 3) + (e - 8));
        const int j = d + n - K;
        __bf16 v = (__bf16)0.0f;
        if (j >= 0 && j < L_) v = kr[j];
        btoep[(size_t)bx * 512 + (size_t)lane * 16 + e] = v;
    }
}

// ---------------------------------------------------------------------------
// K3: Wout fp32 -> bf16 (row-major [v][u]).
// ---------------------------------------------------------------------------
__global__ __launch_bounds__(256)
void wcvt_kernel(const float* __restrict__ Wout, __bf16* __restrict__ wbf)
{
    const int i = blockIdx.x * 256 + threadIdx.x;
    wbf[i] = (__bf16)Wout[i];
}

// ---------------------------------------------------------------------------
// K4: convolution as WMMA GEMM + fused skip + exact GELU.
// Each wave: 4 adjacent 16-wide l-tiles (l0base + 16t), one shared A build
// (u fp32 -> bf16) feeding 4 independent WMMAs per K-chunk. Branch-free body:
// guard tiles absorb the window edges of tiles 0/1. Running pointers, scalar
// trip count (readfirstlane on jmax), prefetch of the 2 new tiles per chunk.
// Epilogue: y += u*D[h]; exact GELU; store bf16 yact[b][h][l].
// ---------------------------------------------------------------------------
__global__ __launch_bounds__(256)
void conv_kernel(const float* __restrict__ u, const __bf16* __restrict__ btoep,
                 const int* __restrict__ jmax, const float* __restrict__ Dp,
                 __bf16* __restrict__ yact)
{
    const int lane   = threadIdx.x & 31;
    const int w      = threadIdx.x >> 5;
    const int h      = blockIdx.y;
    const int l0base = (blockIdx.x * 8 + w) * 64;   // 4 tiles of 16
    const int half   = lane >> 4;
    const int n      = lane & 15;          // A row m = batch; D col = l-offset

    const int jm = __builtin_amdgcn_readfirstlane(jmax[h]);
    int lo = l0base - jm - 31; if (lo < 0) lo = 0;
    const int j0s    = lo & ~31;
    const int j0emax = ((l0base + 63) >> 5) << 5;   // = l0base + 32
    const int iters  = ((j0emax - j0s) >> 5) + 1;

    v8f acc0 = {}, acc1 = {}, acc2 = {}, acc3 = {};
    const float* ap = u + ((size_t)n * H_ + h) * L_ + j0s + 8 * half;
    // lane slice of tile (dbase+t): bp + t*32 (v16bf units); dbase=(l0base-j0)/16
    const int dbase0 = (l0base - j0s) >> 4;
    const v16bf* bp  = (const v16bf*)(btoep
                        + ((size_t)h * NTG + 2 + dbase0) * 512
                        + (size_t)lane * 16);

    for (int it = 0; it < iters; ++it) {
        // shared A operand: u[j0+8h..+7] -> e0..7, u[j0+16+8h..+7] -> e8..15
        const float4 q0 = *(const float4*)(ap);
        const float4 q1 = *(const float4*)(ap + 4);
        const float4 q2 = *(const float4*)(ap + 16);
        const float4 q3 = *(const float4*)(ap + 20);
        // 4 contiguous Toeplitz lane slices (4KB window), clause-friendly
        const v16bf b0 = bp[0];
        const v16bf b1 = bp[32];
        const v16bf b2 = bp[64];
        const v16bf b3 = bp[96];
        // prefetch the 2 tiles that become new ~3 chunks ahead
        __builtin_prefetch((const char*)bp - 6 * 1024, 0, 3);
        __builtin_prefetch((const char*)bp - 5 * 1024, 0, 3);

        v16bf av;
        av[0]=(__bf16)q0.x;  av[1]=(__bf16)q0.y;  av[2]=(__bf16)q0.z;  av[3]=(__bf16)q0.w;
        av[4]=(__bf16)q1.x;  av[5]=(__bf16)q1.y;  av[6]=(__bf16)q1.z;  av[7]=(__bf16)q1.w;
        av[8]=(__bf16)q2.x;  av[9]=(__bf16)q2.y;  av[10]=(__bf16)q2.z; av[11]=(__bf16)q2.w;
        av[12]=(__bf16)q3.x; av[13]=(__bf16)q3.y; av[14]=(__bf16)q3.z; av[15]=(__bf16)q3.w;

        acc0 = __builtin_amdgcn_wmma_f32_16x16x32_bf16(
                   false, av, false, b0, (short)0, acc0, false, false);
        acc1 = __builtin_amdgcn_wmma_f32_16x16x32_bf16(
                   false, av, false, b1, (short)0, acc1, false, false);
        acc2 = __builtin_amdgcn_wmma_f32_16x16x32_bf16(
                   false, av, false, b2, (short)0, acc2, false, false);
        acc3 = __builtin_amdgcn_wmma_f32_16x16x32_bf16(
                   false, av, false, b3, (short)0, acc3, false, false);

        ap += 32;        // next K-chunk of u
        bp -= 64;        // dbase drops by 2 tiles per chunk
    }

    const float Dh = Dp[h];
    v8f accs[4] = { acc0, acc1, acc2, acc3 };
    #pragma unroll
    for (int t = 0; t < 4; ++t) {
        const int lcol = l0base + 16 * t + n;       // D col n = lane%16
        #pragma unroll
        for (int r = 0; r < 8; ++r) {
            const int b    = r + 8 * half;          // D row m = r + 8*(lane/16)
            const float uv = u[((size_t)b * H_ + h) * L_ + lcol];
            const float y  = accs[t][r] + uv * Dh;
            const float g  = 0.5f * y * (1.0f + erff(y * 0.70710678118654752f));
            yact[((size_t)b * H_ + h) * L_ + lcol] = (__bf16)g;
        }
    }
}

// ---------------------------------------------------------------------------
// K5: out[b][v][l] = silu(bout[v] + sum_h W[v][h] * yact[b][h][l]).
// A = Wbf (direct b128 loads per WMMA A layout). B = yact tile staged
// TRANSPOSED in LDS (sYT[l][h], 80B row stride -> 16B-aligned lane slices,
// conflict-free b128 reads), so each lane's B operand is two ds_load_b128.
// Block: 8 waves = 4(v) x 2(l) 16x16 tiles -> 64x32 output, fused bias+SiLU.
// ---------------------------------------------------------------------------
__global__ __launch_bounds__(256)
void gemm2_kernel(const __bf16* __restrict__ yact, const __bf16* __restrict__ wbf,
                  const float* __restrict__ bout, float* __restrict__ out)
{
    __shared__ unsigned short sYT[32][40];   // [l][h]; stride 80B (16B aligned)
    const int tid  = threadIdx.x;
    const int lane = tid & 31, w = tid >> 5;
    const int vt = w & 3, lt = w >> 2;
    const int b   = blockIdx.z;
    const int v0  = blockIdx.y * 64 + vt * 16;
    const int l0b = blockIdx.x * 32;
    const int half = lane >> 4, n = lane & 15;
    const int srow = tid >> 3, scol = (tid & 7) * 4;   // srow: h-offset, scol: l-offset
    const int col  = lt * 16 + n;                      // this lane's l column

    v8f acc = {};
    for (int h0 = 0; h0 < H_; h0 += 32) {
        // stage 32(h) x 32(l) bf16 tile, transposed into sYT[l][h]
        const uint2 g = *(const uint2*)(
            yact + ((size_t)b * H_ + h0 + srow) * L_ + l0b + scol);
        sYT[scol + 0][srow] = (unsigned short)(g.x & 0xffffu);
        sYT[scol + 1][srow] = (unsigned short)(g.x >> 16);
        sYT[scol + 2][srow] = (unsigned short)(g.y & 0xffffu);
        sYT[scol + 3][srow] = (unsigned short)(g.y >> 16);
        __syncthreads();

        // A operand: W row v0+n, K = h0+16*half+e (contiguous)
        const __bf16* wp = wbf + (size_t)(v0 + n) * H_ + h0 + 16 * half;
        union { uint4 q[2]; v16bf v; } au;
        au.q[0] = *(const uint4*)(wp);
        au.q[1] = *(const uint4*)(wp + 8);

        // B operand: lane column `col`, K = 16*half + e -> contiguous in sYT row
        union { uint4 q[2]; v16bf v; } bu;
        const unsigned short* bpl = &sYT[col][16 * half];
        bu.q[0] = *(const uint4*)(bpl);
        bu.q[1] = *(const uint4*)(bpl + 8);

        acc = __builtin_amdgcn_wmma_f32_16x16x32_bf16(
                  false, au.v, false, bu.v, (short)0, acc, false, false);
        __syncthreads();
    }

    #pragma unroll
    for (int r = 0; r < 8; ++r) {
        const int v = v0 + r + 8 * half;
        const int l = l0b + lt * 16 + n;
        const float x = acc[r] + bout[v];
        out[((size_t)b * H_ + v) * L_ + l] = x / (1.0f + expf(-x));  // silu
    }
}

// ---------------------------------------------------------------------------
extern "C" void kernel_launch(void* const* d_in, const int* in_sizes, int n_in,
                              void* d_out, int out_size, void* d_ws, size_t ws_size,
                              hipStream_t stream)
{
    (void)in_sizes; (void)n_in; (void)out_size; (void)ws_size;
    const float* u    = (const float*)d_in[0];
    const float* kern = (const float*)d_in[1];
    const float* Dp   = (const float*)d_in[2];
    const float* Wout = (const float*)d_in[3];
    const float* bo   = (const float*)d_in[4];
    float* out = (float*)d_out;

    char* ws = (char*)d_ws;
    __bf16* ksq   = (__bf16*)(ws + OFF_KSQ);
    int*    jmax  = (int*)   (ws + OFF_JMAX);
    __bf16* btoep = (__bf16*)(ws + OFF_BTOEP);
    __bf16* wbf   = (__bf16*)(ws + OFF_WBF);
    __bf16* yact  = (__bf16*)(ws + OFF_YACT);

    prep_kernel<<<H_, 256, 0, stream>>>(kern, ksq, jmax);
    toep_kernel<<<H_ * NTG, 256, 0, stream>>>(ksq, btoep);
    wcvt_kernel<<<(H_ * H_) / 256, 256, 0, stream>>>(Wout, wbf);
    conv_kernel<<<dim3(L_ / 512, H_), 256, 0, stream>>>(u, btoep, jmax, Dp, yact);
    gemm2_kernel<<<dim3(L_ / 32, H_ / 64, B_), 256, 0, stream>>>(yact, wbf, bo, out);
}
